// ComplexMultiheadAttention_20126216749263
// MI455X (gfx1250) — matrix operand order repaired and checked by reading, not compiled, not run on hardware
//
#include <hip/hip_runtime.h>
#include <hip/hip_bf16.h>

// ---------------------------------------------------------------------------
// Complex multi-head attention for MI455X (gfx1250), wave32 + WMMA f16.
// B=2, N=2048, DIM=256, HEADS=8, DHEAD=32, INNER=256.
// ---------------------------------------------------------------------------

typedef __attribute__((ext_vector_type(16))) _Float16 v16h;
typedef __attribute__((ext_vector_type(8)))  _Float16 v8h;
typedef __attribute__((ext_vector_type(4)))  _Float16 v4h;
typedef __attribute__((ext_vector_type(8)))  float    v8f;

#define NB      2
#define NSEQ    2048
#define DIMX    256
#define NHEADS  8
#define DHEAD   32
#define MROWS   (NB * NSEQ)          // 4096
#define KAUG    (2 * DIMX)           // 512

#define PASS_STRIDE ((long)NB * NHEADS * NSEQ * DHEAD)   // per r-pass in opass

static __device__ inline v8f wmma_f16(v16h a, v16h b, v8f c) {
  // D = A(16x32 f16) * B(32x16 f16) + C(16x16 f32)
  return __builtin_amdgcn_wmma_f32_16x16x32_f16(
      /*neg_a=*/false, a, /*neg_b=*/false, b,
      /*c_mod=*/(short)0, c, /*reuse_a=*/false, /*reuse_b=*/false);
}

// A-fragment (16x32, row-major source, leading dim ld in halves).
// Lane l: row = l&15; elems 0-15 = K [16*(l>>4), +16) contiguous.
static __device__ inline v16h load_a16x32(const _Float16* p, int ld, int lane) {
  const _Float16* q = p + (lane & 15) * ld + 16 * (lane >> 4);
  v8h lo = *(const v8h*)q;
  v8h hi = *(const v8h*)(q + 8);
  v16h a;
#pragma unroll
  for (int i = 0; i < 8; ++i) { a[i] = lo[i]; a[i + 8] = hi[i]; }
  return a;
}

// B-fragment (32x16): source stored as [n][k] row-major (16 contiguous K per lane).
// Lane l: N = l&15, K = 16*(l>>4) + e.
static __device__ inline v16h load_b32x16(const _Float16* p, int ld, int lane) {
  const _Float16* q = p + (lane & 15) * ld + 16 * (lane >> 4);
  return *(const v16h*)q;
}

// ---------------------------------------------------------------------------
// Elementwise prep kernels (all data is L2-resident; these are trivial).
// ---------------------------------------------------------------------------

__global__ void build_xa(const float* __restrict__ xr, const float* __restrict__ xi,
                         _Float16* __restrict__ Xa) {
  int idx = blockIdx.x * 256 + threadIdx.x;          // 4096*512
  int m = idx >> 9, k = idx & 511;
  float v = (k < DIMX) ? xr[m * DIMX + k] : xi[m * DIMX + (k - DIMX)];
  Xa[idx] = (_Float16)v;
}

__global__ void build_wall(const float* __restrict__ Wq_r, const float* __restrict__ Wq_i,
                           const float* __restrict__ Wkv_r, const float* __restrict__ Wkv_i,
                           _Float16* __restrict__ Wall) {
  int idx = blockIdx.x * 256 + threadIdx.x;          // 1536*512
  int out = idx >> 9, k = idx & 511;
  int sec = out >> 8, o = out & 255, kk = k & 255;
  bool hi = (k >= DIMX);
  float v;
  switch (sec) {
    case 0: v = hi ? -Wq_i[o * DIMX + kk]          :  Wq_r[o * DIMX + kk];          break; // q_r
    case 1: v = hi ?  Wq_r[o * DIMX + kk]          :  Wq_i[o * DIMX + kk];          break; // q_i
    case 2: v = hi ? -Wkv_i[o * DIMX + kk]         :  Wkv_r[o * DIMX + kk];         break; // k_r
    case 3: v = hi ?  Wkv_r[o * DIMX + kk]         :  Wkv_i[o * DIMX + kk];         break; // k_i
    case 4: v = hi ? -Wkv_i[(o + 256) * DIMX + kk] :  Wkv_r[(o + 256) * DIMX + kk]; break; // v_r
    default:v = hi ?  Wkv_r[(o + 256) * DIMX + kk] :  Wkv_i[(o + 256) * DIMX + kk]; break; // v_i
  }
  Wall[idx] = (_Float16)v;
}

__global__ void build_woa(const float* __restrict__ Wo_r, const float* __restrict__ Wo_i,
                          _Float16* __restrict__ Woa) {
  int idx = blockIdx.x * 256 + threadIdx.x;          // 512*512
  int out = idx >> 9, k = idx & 511;
  int ri = out >> 8, o = out & 255, kk = k & 255;
  bool hi = (k >= DIMX);
  float v = (ri == 0) ? (hi ? -Wo_i[o * DIMX + kk] : Wo_r[o * DIMX + kk])
                      : (hi ?  Wo_r[o * DIMX + kk] : Wo_i[o * DIMX + kk]);
  Woa[idx] = (_Float16)v;
}

// Combine the four attention passes (o_r = rr-ii, o_i = ri+ir), merge heads,
// and build the augmented f16 input [4096][512] for the output projection.
__global__ void build_oaug(const float* __restrict__ opass, _Float16* __restrict__ Oaug) {
  int idx = blockIdx.x * 256 + threadIdx.x;          // 4096*512
  int m = idx >> 9, k = idx & 511;
  int hd = k & 255, h = hd >> 5, d = hd & 31;
  int b = m >> 11, n = m & 2047;
  long base = ((((long)b * NHEADS) + h) * NSEQ + n) * DHEAD + d;
  float v;
  if (k < DIMX) v = opass[0 * PASS_STRIDE + base] - opass[3 * PASS_STRIDE + base];
  else          v = opass[1 * PASS_STRIDE + base] + opass[2 * PASS_STRIDE + base];
  Oaug[idx] = (_Float16)v;
}

// ---------------------------------------------------------------------------
// GEMM 1: Y(4096 x 1536) = Xa(4096 x 512) * Wall^T.
// Each wave: 32x64 tile, register double-buffered over K so loads for step
// k+1 are in flight while the 8 WMMAs of step k execute.
// Output scattered into qkv buffer [tensor][b][h][n][d] in f16.
// ---------------------------------------------------------------------------
__global__ __launch_bounds__(256) void gemm_qkv(const _Float16* __restrict__ Xa,
                                                const _Float16* __restrict__ Wall,
                                                _Float16* __restrict__ qkv) {
  int w    = blockIdx.x * 8 + (threadIdx.x >> 5);
  int lane = threadIdx.x & 31;
  int mtile = w & 127;          // 128 row tiles of 32
  int ngrp  = w >> 7;           // 24 column groups of 64
  int row0  = mtile * 32;
  const _Float16* Arow0 = Xa + (long)row0 * KAUG;
  const _Float16* Arow1 = Xa + (long)(row0 + 16) * KAUG;
  const _Float16* Wg    = Wall + (long)(ngrp * 64) * KAUG;

  v8f c[2][4];
#pragma unroll
  for (int t = 0; t < 2; ++t)
#pragma unroll
    for (int j = 0; j < 4; ++j) c[t][j] = (v8f){};

  v16h a0 = load_a16x32(Arow0, KAUG, lane);
  v16h a1 = load_a16x32(Arow1, KAUG, lane);
  v16h b0 = load_b32x16(Wg +  0 * KAUG, KAUG, lane);
  v16h b1 = load_b32x16(Wg + 16 * KAUG, KAUG, lane);
  v16h b2 = load_b32x16(Wg + 32 * KAUG, KAUG, lane);
  v16h b3 = load_b32x16(Wg + 48 * KAUG, KAUG, lane);

  for (int k0 = 32; k0 < KAUG; k0 += 32) {
    v16h a0n = load_a16x32(Arow0 + k0, KAUG, lane);
    v16h a1n = load_a16x32(Arow1 + k0, KAUG, lane);
    v16h b0n = load_b32x16(Wg +  0 * KAUG + k0, KAUG, lane);
    v16h b1n = load_b32x16(Wg + 16 * KAUG + k0, KAUG, lane);
    v16h b2n = load_b32x16(Wg + 32 * KAUG + k0, KAUG, lane);
    v16h b3n = load_b32x16(Wg + 48 * KAUG + k0, KAUG, lane);
    c[0][0] = wmma_f16(a0, b0, c[0][0]);
    c[0][1] = wmma_f16(a0, b1, c[0][1]);
    c[0][2] = wmma_f16(a0, b2, c[0][2]);
    c[0][3] = wmma_f16(a0, b3, c[0][3]);
    c[1][0] = wmma_f16(a1, b0, c[1][0]);
    c[1][1] = wmma_f16(a1, b1, c[1][1]);
    c[1][2] = wmma_f16(a1, b2, c[1][2]);
    c[1][3] = wmma_f16(a1, b3, c[1][3]);
    a0 = a0n; a1 = a1n; b0 = b0n; b1 = b1n; b2 = b2n; b3 = b3n;
  }
  c[0][0] = wmma_f16(a0, b0, c[0][0]);
  c[0][1] = wmma_f16(a0, b1, c[0][1]);
  c[0][2] = wmma_f16(a0, b2, c[0][2]);
  c[0][3] = wmma_f16(a0, b3, c[0][3]);
  c[1][0] = wmma_f16(a1, b0, c[1][0]);
  c[1][1] = wmma_f16(a1, b1, c[1][1]);
  c[1][2] = wmma_f16(a1, b2, c[1][2]);
  c[1][3] = wmma_f16(a1, b3, c[1][3]);

  int half = lane >> 4, coln = lane & 15;
#pragma unroll
  for (int t = 0; t < 2; ++t) {
#pragma unroll
    for (int j = 0; j < 4; ++j) {
#pragma unroll
      for (int g = 0; g < 8; ++g) {
        int cidx = ngrp * 64 + j * 16 + coln;
        int m = row0 + t * 16 + g + 8 * half;
        int tt = cidx >> 8, ct = cidx & 255, h = ct >> 5, d = ct & 31;
        int b = m >> 11, n = m & 2047;
        qkv[(((((long)tt * NB + b) * NHEADS + h) * NSEQ) + n) * DHEAD + d] =
            (_Float16)c[t][j][g];
      }
    }
  }
}

// ---------------------------------------------------------------------------
// Flash attention: one workgroup per (r, b, h, 128-row block); 8 waves, each
// owns 16 query rows; K/V blocks of 32 keys staged cooperatively in LDS with
// the next block's global loads prefetched into registers during compute.
// ---------------------------------------------------------------------------
__global__ __launch_bounds__(256) void attn_flash(const _Float16* __restrict__ qkv,
                                                  float* __restrict__ opass) {
  __shared__ _Float16 Klds[32 * 40];      // [key][d], padded
  __shared__ _Float16 Vt[32 * 40];        // [d][key], padded (transposed at staging)
  __shared__ _Float16 Pl[8 * 16 * 40];    // per-wave P scratch [row][key]

  int wg = blockIdx.x;
  int rowblk = wg & 15;
  int h = (wg >> 4) & 7;
  int b = (wg >> 7) & 1;
  int r = (wg >> 8) & 3;

  int wave = threadIdx.x >> 5;
  int lane = threadIdx.x & 31;
  int half = lane >> 4, coln = lane & 15;

  int t_q = (r >= 2) ? 1 : 0;
  int t_k = (r & 1)  ? 3 : 2;
  int t_v = (r >= 2) ? 5 : 4;
  // sign of the (-k_i) passes folded into the softmax scale
  float scl = ((r & 1) ? -1.0f : 1.0f) * 0.17677669529663687f;  // +-1/sqrt(32)

  const _Float16* qbase = qkv + ((((long)t_q * NB + b) * NHEADS + h) * NSEQ) * DHEAD;
  const _Float16* kbase = qkv + ((((long)t_k * NB + b) * NHEADS + h) * NSEQ) * DHEAD;
  const _Float16* vbase = qkv + ((((long)t_v * NB + b) * NHEADS + h) * NSEQ) * DHEAD;

  int qrow0 = rowblk * 128 + wave * 16;
  v16h qa = load_a16x32(qbase + (long)qrow0 * DHEAD, DHEAD, lane);

  float m_run[8], l_run[8];
  v8f acc0 = {}, acc1 = {};
#pragma unroll
  for (int g = 0; g < 8; ++g) { m_run[g] = -1.0e30f; l_run[g] = 0.0f; }

  // staging coordinates for this thread: 4 halves of one (key,d4) slot
  int skey = threadIdx.x >> 3;
  int sd4  = (threadIdx.x & 7) * 4;
  const int NKB = NSEQ / 32;

  // prefetch block 0 into registers
  v4h kreg = *(const v4h*)(kbase + ((long)0 * 32 + skey) * DHEAD + sd4);
  v4h vreg = *(const v4h*)(vbase + ((long)0 * 32 + skey) * DHEAD + sd4);

  for (int kb = 0; kb < NKB; ++kb) {
    __syncthreads();                        // LDS free from previous block
    *(v4h*)&Klds[skey * 40 + sd4] = kreg;
#pragma unroll
    for (int i = 0; i < 4; ++i) Vt[(sd4 + i) * 40 + skey] = vreg[i];

    // prefetch next block (clamped, branchless) while this block computes
    int kbn = (kb + 1 < NKB) ? (kb + 1) : kb;
    v4h kregn = *(const v4h*)(kbase + ((long)kbn * 32 + skey) * DHEAD + sd4);
    v4h vregn = *(const v4h*)(vbase + ((long)kbn * 32 + skey) * DHEAD + sd4);
    __syncthreads();                        // staging visible to all waves

    // S = Q * K^T for this 32-key block (two 16-key WMMA tiles)
    v16h kb0 = load_b32x16(Klds, 40, lane);            // keys 0-15
    v16h kb1 = load_b32x16(Klds + 16 * 40, 40, lane);  // keys 16-31
    v8f s0t = {}, s1t = {};
    s0t = wmma_f16(qa, kb0, s0t);
    s1t = wmma_f16(qa, kb1, s1t);

    float s0[8], s1[8], mx[8];
#pragma unroll
    for (int g = 0; g < 8; ++g) {
      s0[g] = s0t[g] * scl;
      s1[g] = s1t[g] * scl;
      mx[g] = fmaxf(s0[g], s1[g]);
    }
    // row max across the 16 lanes of this half (rows live in one half)
#pragma unroll
    for (int msk = 1; msk <= 8; msk <<= 1)
#pragma unroll
      for (int g = 0; g < 8; ++g) mx[g] = fmaxf(mx[g], __shfl_xor(mx[g], msk, 32));

    float mnew[8], al[8], p0[8], p1[8], rs[8];
#pragma unroll
    for (int g = 0; g < 8; ++g) {
      mnew[g] = fmaxf(m_run[g], mx[g]);
      al[g]   = __expf(m_run[g] - mnew[g]);
      p0[g]   = __expf(s0[g] - mnew[g]);
      p1[g]   = __expf(s1[g] - mnew[g]);
      rs[g]   = p0[g] + p1[g];
    }
#pragma unroll
    for (int msk = 1; msk <= 8; msk <<= 1)
#pragma unroll
      for (int g = 0; g < 8; ++g) rs[g] += __shfl_xor(rs[g], msk, 32);
#pragma unroll
    for (int g = 0; g < 8; ++g) {
      l_run[g] = l_run[g] * al[g] + rs[g];
      m_run[g] = mnew[g];
      acc0[g] *= al[g];
      acc1[g] *= al[g];
    }

    // C-layout -> A-layout conversion of P through per-wave LDS scratch
    _Float16* pw = Pl + wave * (16 * 40);
#pragma unroll
    for (int g = 0; g < 8; ++g) {
      pw[(g + 8 * half) * 40 + coln]      = (_Float16)p0[g];
      pw[(g + 8 * half) * 40 + 16 + coln] = (_Float16)p1[g];
    }
    __builtin_amdgcn_wave_barrier();     // wave-internal LDS RAW ordering
    v16h pa = load_a16x32(pw, 40, lane);

    // O += P * V (two 16-wide d tiles; Vt is [d][key] so B loads are contiguous)
    v16h vb0 = load_b32x16(Vt, 40, lane);            // d 0-15
    v16h vb1 = load_b32x16(Vt + 16 * 40, 40, lane);  // d 16-31
    acc0 = wmma_f16(pa, vb0, acc0);
    acc1 = wmma_f16(pa, vb1, acc1);

    kreg = kregn; vreg = vregn;
  }

  // normalize and store this pass's output [r][b][h][n][d] in f32
  float* op = opass + ((((long)r * NB + b) * NHEADS + h) * NSEQ + qrow0) * DHEAD;
#pragma unroll
  for (int g = 0; g < 8; ++g) {
    int row = g + 8 * half;
    float inv = 1.0f / l_run[g];
    op[row * DHEAD + coln]      = acc0[g] * inv;
    op[row * DHEAD + 16 + coln] = acc1[g] * inv;
  }
}

// ---------------------------------------------------------------------------
// GEMM 2: Y(4096 x 512) = Oaug(4096 x 512) * Woa^T, 32x64 tile per wave,
// register double-buffered; output interleaved as (B, N, DIM, 2) f32.
// ---------------------------------------------------------------------------
__global__ __launch_bounds__(256) void gemm_out(const _Float16* __restrict__ Oaug,
                                                const _Float16* __restrict__ Woa,
                                                float* __restrict__ out) {
  int w    = blockIdx.x * 8 + (threadIdx.x >> 5);
  int lane = threadIdx.x & 31;
  int mtile = w & 127;          // 128 row tiles of 32
  int ngrp  = w >> 7;           // 8 column groups of 64
  int row0  = mtile * 32;
  const _Float16* Arow0 = Oaug + (long)row0 * KAUG;
  const _Float16* Arow1 = Oaug + (long)(row0 + 16) * KAUG;
  const _Float16* Wg    = Woa + (long)(ngrp * 64) * KAUG;

  v8f c[2][4];
#pragma unroll
  for (int t = 0; t < 2; ++t)
#pragma unroll
    for (int j = 0; j < 4; ++j) c[t][j] = (v8f){};

  v16h a0 = load_a16x32(Arow0, KAUG, lane);
  v16h a1 = load_a16x32(Arow1, KAUG, lane);
  v16h b0 = load_b32x16(Wg +  0 * KAUG, KAUG, lane);
  v16h b1 = load_b32x16(Wg + 16 * KAUG, KAUG, lane);
  v16h b2 = load_b32x16(Wg + 32 * KAUG, KAUG, lane);
  v16h b3 = load_b32x16(Wg + 48 * KAUG, KAUG, lane);

  for (int k0 = 32; k0 < KAUG; k0 += 32) {
    v16h a0n = load_a16x32(Arow0 + k0, KAUG, lane);
    v16h a1n = load_a16x32(Arow1 + k0, KAUG, lane);
    v16h b0n = load_b32x16(Wg +  0 * KAUG + k0, KAUG, lane);
    v16h b1n = load_b32x16(Wg + 16 * KAUG + k0, KAUG, lane);
    v16h b2n = load_b32x16(Wg + 32 * KAUG + k0, KAUG, lane);
    v16h b3n = load_b32x16(Wg + 48 * KAUG + k0, KAUG, lane);
    c[0][0] = wmma_f16(a0, b0, c[0][0]);
    c[0][1] = wmma_f16(a0, b1, c[0][1]);
    c[0][2] = wmma_f16(a0, b2, c[0][2]);
    c[0][3] = wmma_f16(a0, b3, c[0][3]);
    c[1][0] = wmma_f16(a1, b0, c[1][0]);
    c[1][1] = wmma_f16(a1, b1, c[1][1]);
    c[1][2] = wmma_f16(a1, b2, c[1][2]);
    c[1][3] = wmma_f16(a1, b3, c[1][3]);
    a0 = a0n; a1 = a1n; b0 = b0n; b1 = b1n; b2 = b2n; b3 = b3n;
  }
  c[0][0] = wmma_f16(a0, b0, c[0][0]);
  c[0][1] = wmma_f16(a0, b1, c[0][1]);
  c[0][2] = wmma_f16(a0, b2, c[0][2]);
  c[0][3] = wmma_f16(a0, b3, c[0][3]);
  c[1][0] = wmma_f16(a1, b0, c[1][0]);
  c[1][1] = wmma_f16(a1, b1, c[1][1]);
  c[1][2] = wmma_f16(a1, b2, c[1][2]);
  c[1][3] = wmma_f16(a1, b3, c[1][3]);

  int half = lane >> 4, coln = lane & 15;
#pragma unroll
  for (int t = 0; t < 2; ++t) {
#pragma unroll
    for (int j = 0; j < 4; ++j) {
#pragma unroll
      for (int g = 0; g < 8; ++g) {
        int cidx = ngrp * 64 + j * 16 + coln;
        int m = row0 + t * 16 + g + 8 * half;
        int dim = cidx & 255, ri = cidx >> 8;
        out[((long)m * DIMX + dim) * 2 + ri] = c[t][j][g];
      }
    }
  }
}

// ---------------------------------------------------------------------------
extern "C" void kernel_launch(void* const* d_in, const int* in_sizes, int n_in,
                              void* d_out, int out_size, void* d_ws, size_t ws_size,
                              hipStream_t stream) {
  (void)in_sizes; (void)n_in; (void)out_size; (void)ws_size;
  const float* x_r   = (const float*)d_in[0];
  const float* x_i   = (const float*)d_in[1];
  const float* Wq_r  = (const float*)d_in[2];
  const float* Wq_i  = (const float*)d_in[3];
  const float* Wkv_r = (const float*)d_in[4];
  const float* Wkv_i = (const float*)d_in[5];
  const float* Wo_r  = (const float*)d_in[6];
  const float* Wo_i  = (const float*)d_in[7];
  float* out = (float*)d_out;

  char* ws = (char*)d_ws;
  _Float16* Xa    = (_Float16*)(ws);                       //  4,194,304 B
  _Float16* Wall  = (_Float16*)(ws + 4194304);             //  1,572,864 B
  _Float16* Woa   = (_Float16*)(ws + 5767168);             //    524,288 B
  _Float16* qkv   = (_Float16*)(ws + 6291456);             // 12,582,912 B
  float*    opass = (float*)   (ws + 18874368);            // 16,777,216 B
  _Float16* Oaug  = (_Float16*)(ws + 35651584);            //  4,194,304 B

  build_xa  <<<8192, 256, 0, stream>>>(x_r, x_i, Xa);
  build_wall<<<3072, 256, 0, stream>>>(Wq_r, Wq_i, Wkv_r, Wkv_i, Wall);
  build_woa <<<1024, 256, 0, stream>>>(Wo_r, Wo_i, Woa);

  gemm_qkv  <<<384, 256, 0, stream>>>(Xa, Wall, qkv);      // 24 grp * 128 mtiles / 8 waves
  attn_flash<<<1024, 256, 0, stream>>>(qkv, opass);        // 4*2*8*16 workgroups
  build_oaug<<<8192, 256, 0, stream>>>(opass, Oaug);
  gemm_out  <<<128, 256, 0, stream>>>(Oaug, Woa, out);     // 8 grp * 128 mtiles / 8 waves
}
